// Attention_13056700580138
// MI455X (gfx1250) — matrix operand order repaired and checked by reading, not compiled, not run on hardware
//
#include <hip/hip_runtime.h>
#include <hip/hip_bf16.h>
#include <math.h>

// ---------------------------------------------------------------------------
// SAGAN attention block, fused flash-attention style for MI455X (gfx1250).
//   B=8, H=W=64 -> N=4096, C=256, CK=32
// Stage 0: weights -> bf16, transposed [Cout][Cin]   (WMMA B-operand layout)
// Stage 1: projections f,g (bf16 [32768][32]) and hh channel-major bf16 [8][256][4096]
// Stage 2: per-wave flash attention (16 queries); per-BLOCK double-buffered
//          staging of the shared K/V slabs into LDS:
//            V slab (256x32 bf16, 16KB): Tensor Data Mover (tensor_load_to_lds)
//            K slab (32x32 bf16, 2KB):   global_load_async_to_lds_b128
//          Fused o@Wo + bias + gamma*residual epilogue.
// All matmuls via v_wmma_f32_16x16x32_bf16 (wave32).
// ---------------------------------------------------------------------------

typedef __bf16 bf16_t;
typedef __attribute__((ext_vector_type(16))) __bf16 v16bf;
typedef __attribute__((ext_vector_type(8)))  __bf16 v8bf;
typedef __attribute__((ext_vector_type(8)))  float  v8f;
typedef __attribute__((ext_vector_type(4)))  unsigned int v4u;
typedef __attribute__((ext_vector_type(4)))  int v4i;
typedef __attribute__((ext_vector_type(8)))  int v8i;

#define WMMA_BF16(A, B, C) \
  __builtin_amdgcn_wmma_f32_16x16x32_bf16(false, (A), false, (B), (short)0, (C), false, false)

#if __has_builtin(__builtin_amdgcn_tensor_load_to_lds)
#define HAVE_TDM 1
#else
#define HAVE_TDM 0
#endif

static __device__ __forceinline__ void s_wait_dscnt0() {
  asm volatile("s_wait_dscnt 0" ::: "memory");
}
static __device__ __forceinline__ void s_wait_asynccnt0() {
  asm volatile("s_wait_asynccnt 0" ::: "memory");
}
static __device__ __forceinline__ void s_wait_asynccnt1() {
  asm volatile("s_wait_asynccnt 1" ::: "memory");
}

// Generic->LDS offset: flat LDS addresses carry the 32-bit LDS offset in the
// low DWORD (aperture occupies the high DWORD), so truncation recovers it.
static __device__ __forceinline__ unsigned lds_off_u32(const void* p) {
  return (unsigned)(unsigned long long)p;
}

// Async copy of one 16B chunk global -> LDS (ASYNCcnt-tracked).
static __device__ __forceinline__ void async_ld_b128(unsigned ldsoff, const void* g) {
  asm volatile("global_load_async_to_lds_b128 %0, %1, off"
               :: "v"(ldsoff), "v"(g) : "memory");
}

// TDM: 2D tile, 256 rows (stride 4096 elements) x 32 contiguous bf16 -> LDS.
static __device__ __forceinline__ void tdm_load_v_slab(unsigned lds_addr, const void* gaddr) {
#if HAVE_TDM
  unsigned long long ga = (unsigned long long)gaddr;
  v4u g0;
  g0[0] = 1u;                                               // count=1, user mode
  g0[1] = lds_addr;                                         // lds_addr
  g0[2] = (unsigned)ga;                                     // global_addr[31:0]
  g0[3] = (unsigned)((ga >> 32) & 0x01FFFFFFu) | (2u << 30);// global_addr[56:32], type=2
  v8i g1;
  g1[0] = (int)(1u << 16);      // workgroup_mask=0, data_size=1 (2B), no pad/iter
  g1[1] = (int)(4096u << 16);   // tensor_dim0 = 4096 (low16 at bits 63:48)
  g1[2] = (int)(256u << 16);    // tensor_dim0 hi=0; tensor_dim1 = 256 (low16)
  g1[3] = (int)(32u << 16);     // tensor_dim1 hi=0; tile_dim0 = 32
  g1[4] = 256;                  // tile_dim1 = 256; tile_dim2 = 0
  g1[5] = 4096;                 // tensor_dim0_stride = 4096 (low32)
  g1[6] = 0;                    // stride hi / tensor_dim1_stride lo
  g1[7] = 0;
  v4i z4 = {};
#if defined(__clang_major__) && (__clang_major__ >= 23)
  v8i z8 = {};
  __builtin_amdgcn_tensor_load_to_lds(g0, g1, z4, z4, z8, 0);
#else
  __builtin_amdgcn_tensor_load_to_lds(g0, g1, z4, z4, 0);
#endif
#else
  (void)lds_addr; (void)gaddr;
#endif
}

// ---------------------------------------------------------------------------
// Stage 0: src[rows][cols] f32  ->  dst[cols][rows] bf16
// ---------------------------------------------------------------------------
__global__ void k_transpose_bf16(const float* __restrict__ src,
                                 bf16_t* __restrict__ dst, int rows, int cols) {
  int i = blockIdx.x * blockDim.x + threadIdx.x;
  if (i < rows * cols) {
    int r = i / cols, c = i % cols;
    dst[(size_t)c * rows + r] = (bf16_t)src[i];
  }
}

// ---------------------------------------------------------------------------
// Stage 1: projections. One wave = one 16x16 output tile.
// Output tiles: 2048 row tiles x 20 col tiles (f:0-1, g:2-3, hh:4-19)
// ---------------------------------------------------------------------------
__global__ __launch_bounds__(128) void k_proj(
    const float*  __restrict__ x,    // [32768][256]
    const float*  __restrict__ bf,
    const float*  __restrict__ bg,
    const float*  __restrict__ bh,
    const bf16_t* __restrict__ WfT,  // [32][256]
    const bf16_t* __restrict__ WgT,  // [32][256]
    const bf16_t* __restrict__ WhT,  // [256][256]
    bf16_t* __restrict__ fB,         // [32768][32]
    bf16_t* __restrict__ gB,         // [32768][32]
    bf16_t* __restrict__ hhT)        // [8][256][4096]
{
  const int lane = threadIdx.x & 31;
  const int wid  = blockIdx.x * 4 + (threadIdx.x >> 5);
  const int rt = wid / 20, ct = wid % 20;
  const int p0 = rt * 16;
  const int n  = lane & 15;   // A-row index AND B-column index for this lane
  const int kh = lane >> 4;   // lane half selects contiguous K sub-range 16*kh..16*kh+15

  const bf16_t* wt;
  int c0;
  if (ct < 2)      { wt = WfT; c0 = ct * 16; }
  else if (ct < 4) { wt = WgT; c0 = (ct - 2) * 16; }
  else             { wt = WhT; c0 = (ct - 4) * 16; }

  const float*  arow = x  + (size_t)(p0 + n) * 256;
  const bf16_t* brow = wt + (size_t)(c0 + n) * 256;

  v8f acc = {};
  #pragma unroll
  for (int k = 0; k < 8; ++k) {
    const int k0 = k * 32;
    // A tile (16x32): lane holds 16 contiguous K starting at kh*16
    v8f ch0 = *(const v8f*)(arow + k0 + kh * 16);
    v8f ch1 = *(const v8f*)(arow + k0 + kh * 16 + 8);
    v16bf av;
    #pragma unroll
    for (int i = 0; i < 8; ++i) { av[i] = (bf16_t)ch0[i]; av[i + 8] = (bf16_t)ch1[i]; }
    // B tile (32x16): lane = column, 16 contiguous K starting at kh*16
    v16bf bv = *(const v16bf*)(brow + k0 + kh * 16);
    acc = WMMA_BF16(av, bv, acc);
  }

  // D layout: lane holds (m = v + 8*kh, n = lane&15)
  if (ct < 4) {
    const float* bias = (ct < 2) ? bf : bg;
    bf16_t* dst = (ct < 2) ? fB : gB;
    float bval = bias[c0 + n];
    #pragma unroll
    for (int v = 0; v < 8; ++v) {
      int m = v + kh * 8;
      dst[(size_t)(p0 + m) * 32 + (c0 + n)] = (bf16_t)(acc[v] + bval);
    }
  } else {
    float bval = bh[c0 + n];
    int b  = p0 >> 12;        // batch (tiles never straddle: 16 | 4096)
    int pl = p0 & 4095;       // pixel within batch
    bf16_t tmp[8];
    #pragma unroll
    for (int v = 0; v < 8; ++v) tmp[v] = (bf16_t)(acc[v] + bval);
    // channel-major: pixels are contiguous -> one 16B store per lane
    bf16_t* dst = hhT + ((size_t)(b * 256 + c0 + n) * 4096 + pl + kh * 8);
    *(uint4*)dst = *(const uint4*)tmp;
  }
}

// ---------------------------------------------------------------------------
// Stage 2: flash attention + output projection + residual.
// One wave = 16 query rows; 4 waves/block share one batch's K/V stream.
// 128 key blocks of 32 keys, double-buffered TDM (V) + async-LDS (K) staging.
// ---------------------------------------------------------------------------
__global__ __launch_bounds__(128) void k_attn(
    const float*  __restrict__ x,     // [32768][256]
    const bf16_t* __restrict__ fB,    // keys   [32768][32]
    const bf16_t* __restrict__ gB,    // query  [32768][32]
    const bf16_t* __restrict__ hhT,   // values [8][256][4096] channel-major
    const bf16_t* __restrict__ WoT,   // [256][256]
    const float*  __restrict__ bo,    // [256]
    const float*  __restrict__ gamma, // [1]
    float* __restrict__ out)          // [32768][256]
{
  __shared__ __align__(64) bf16_t Vst[2][256][32];  // V slab double buffer (32KB)
  __shared__ __align__(64) bf16_t Kst[2][32][32];   // K slab double buffer (4KB)
  __shared__ __align__(16) bf16_t Pst[4][16][32];   // P tile staging (per wave)
  __shared__ __align__(16) bf16_t Ost[4][16][256];  // O tile staging (per wave)

  const int lane = threadIdx.x & 31;
  const int w    = threadIdx.x >> 5;
  const int wid  = blockIdx.x * 4 + w;
  const int b    = wid >> 8;          // uniform across the block (4 | 256)
  const int q0   = (wid & 255) * 16;
  const int n    = lane & 15;
  const int kh   = lane >> 4;

  const bf16_t* fbase = fB  + (size_t)b * 4096 * 32;
  const bf16_t* vbase = hhT + (size_t)b * 256 * 4096;

  // Q tile in A layout (CK == 32 == one WMMA K step)
  const bf16_t* qrow = gB + (size_t)(b * 4096 + q0 + n) * 32;
  v16bf qa;
  *(v8bf*)&qa         = *(const v8bf*)(qrow + kh * 16);
  *(((v8bf*)&qa) + 1) = *(const v8bf*)(qrow + kh * 16 + 8);

  v8f zero = {};
  v8f oacc[16];
  #pragma unroll
  for (int t = 0; t < 16; ++t) oacc[t] = zero;

  float mrow[8], lrow[8];
  #pragma unroll
  for (int v = 0; v < 8; ++v) { mrow[v] = -INFINITY; lrow[v] = 0.0f; }

  // ---- staging helpers (key0n = first key of the block to stage) ----
  auto stage_v = [&](int buf, int key0n) {
#if HAVE_TDM
    if (w == 0) tdm_load_v_slab(lds_off_u32(&Vst[buf][0][0]), vbase + key0n);
#else
    // fallback: cooperative copy, 128 threads x 2 channel rows x 64B
    #pragma unroll
    for (int r = 0; r < 2; ++r) {
      int ch = threadIdx.x * 2 + r;
      const uint4* g = (const uint4*)(vbase + (size_t)ch * 4096 + key0n);
      uint4* l = (uint4*)&Vst[buf][ch][0];
      l[0] = g[0]; l[1] = g[1]; l[2] = g[2]; l[3] = g[3];
    }
#endif
  };
  auto stage_k = [&](int buf, int key0n) {
    int r  = threadIdx.x >> 2;   // key row 0..31
    int cc = threadIdx.x & 3;    // 16B chunk 0..3
    const bf16_t* g = fbase + (size_t)(key0n + r) * 32 + cc * 8;
    async_ld_b128(lds_off_u32(&Kst[buf][r][cc * 8]), g);
  };

  // prologue: stage block 0
  stage_v(0, 0);
  stage_k(0, 0);

  int cur = 0;
  for (int kb = 0; kb < 128; ++kb) {
    const int key0 = kb * 32;

    // issue next block's DMA, then wait for the current block
    if (kb + 1 < 128) {
      stage_v(cur ^ 1, key0 + 32);
      stage_k(cur ^ 1, key0 + 32);
      s_wait_asynccnt1();                          // K slab kb done (1 in flight)
#if HAVE_TDM
      if (w == 0) __builtin_amdgcn_s_wait_tensorcnt((short)1);
#endif
    } else {
      s_wait_asynccnt0();
#if HAVE_TDM
      if (w == 0) __builtin_amdgcn_s_wait_tensorcnt((short)0);
#endif
    }
    __syncthreads();   // publish Kst[cur] / Vst[cur] to all waves

    // K^T B-tiles from LDS: lane = key column, 16 contiguous channels
    v16bf ktb0 = *(const v16bf*)&Kst[cur][n][kh * 16];
    v16bf ktb1 = *(const v16bf*)&Kst[cur][16 + n][kh * 16];

    v8f s0 = WMMA_BF16(qa, ktb0, zero);   // scores, keys key0..key0+15
    v8f s1 = WMMA_BF16(qa, ktb1, zero);   // scores, keys key0+16..key0+31

    // ---- online softmax (row = v + 8*kh, spread over 16 lanes n=0..15) ----
    float corr[8];
    #pragma unroll
    for (int v = 0; v < 8; ++v) {
      float mx = fmaxf(s0[v], s1[v]);
      #pragma unroll
      for (int d = 8; d >= 1; d >>= 1) mx = fmaxf(mx, __shfl_xor(mx, d, 32));
      float mnew = fmaxf(mrow[v], mx);
      float p0v = __expf(s0[v] - mnew);
      float p1v = __expf(s1[v] - mnew);
      float ls  = p0v + p1v;
      #pragma unroll
      for (int d = 8; d >= 1; d >>= 1) ls += __shfl_xor(ls, d, 32);
      float c = __expf(mrow[v] - mnew);   // first iter: exp(-inf)=0
      lrow[v] = lrow[v] * c + ls;
      mrow[v] = mnew;
      corr[v] = c;
      s0[v] = p0v;
      s1[v] = p1v;
    }
    #pragma unroll
    for (int t = 0; t < 16; ++t) {
      #pragma unroll
      for (int v = 0; v < 8; ++v) oacc[t][v] *= corr[v];
    }

    // ---- D-layout -> A-layout for P via LDS ----
    #pragma unroll
    for (int v = 0; v < 8; ++v) {
      int m = v + kh * 8;
      Pst[w][m][n]      = (bf16_t)s0[v];
      Pst[w][m][16 + n] = (bf16_t)s1[v];
    }
    s_wait_dscnt0();
    v16bf pa;
    *(v8bf*)&pa         = *(const v8bf*)&Pst[w][n][kh * 16];
    *(((v8bf*)&pa) + 1) = *(const v8bf*)&Pst[w][n][kh * 16 + 8];

    // ---- O += P @ V : 16 channel tiles from the LDS V slab, K = 32 keys ----
    #pragma unroll
    for (int t = 0; t < 16; ++t) {
      v16bf vb = *(const v16bf*)&Vst[cur][t * 16 + n][kh * 16];
      oacc[t] = WMMA_BF16(pa, vb, oacc[t]);
    }

    __syncthreads();   // all waves done reading buffers before kb+2 overwrites
    cur ^= 1;
  }

  // ---- epilogue: normalize, stage O, fused o@Wo + bias + gamma*residual ----
  #pragma unroll
  for (int t = 0; t < 16; ++t) {
    #pragma unroll
    for (int v = 0; v < 8; ++v) {
      float o = oacc[t][v] / lrow[v];
      Ost[w][v + kh * 8][t * 16 + n] = (bf16_t)o;
    }
  }
  s_wait_dscnt0();

  const float gam = gamma[0];
  const float* xrow = x   + (size_t)(b * 4096 + q0) * 256;
  float*       orow = out + (size_t)(b * 4096 + q0) * 256;

  #pragma unroll 1
  for (int ct = 0; ct < 16; ++ct) {
    v8f acc = zero;
    const bf16_t* wrow = WoT + (size_t)(ct * 16 + n) * 256;
    #pragma unroll
    for (int k = 0; k < 8; ++k) {
      v16bf av;
      *(v8bf*)&av         = *(const v8bf*)&Ost[w][n][k * 32 + kh * 16];
      *(((v8bf*)&av) + 1) = *(const v8bf*)&Ost[w][n][k * 32 + kh * 16 + 8];
      v16bf bv = *(const v16bf*)(wrow + k * 32 + kh * 16);
      acc = WMMA_BF16(av, bv, acc);
    }
    float bov = bo[ct * 16 + n];
    #pragma unroll
    for (int v = 0; v < 8; ++v) {
      int m = v + kh * 8;
      size_t idx = (size_t)m * 256 + ct * 16 + n;
      orow[idx] = gam * (acc[v] + bov) + xrow[idx];
    }
  }
}

// ---------------------------------------------------------------------------
extern "C" void kernel_launch(void* const* d_in, const int* in_sizes, int n_in,
                              void* d_out, int out_size, void* d_ws, size_t ws_size,
                              hipStream_t stream) {
  (void)in_sizes; (void)n_in; (void)out_size; (void)ws_size;

  const float* x     = (const float*)d_in[0];
  const float* Wf    = (const float*)d_in[1];
  const float* bf    = (const float*)d_in[2];
  const float* Wg    = (const float*)d_in[3];
  const float* bg    = (const float*)d_in[4];
  const float* Wh    = (const float*)d_in[5];
  const float* bh    = (const float*)d_in[6];
  const float* Wo    = (const float*)d_in[7];
  const float* bo    = (const float*)d_in[8];
  const float* gamma = (const float*)d_in[9];
  float* out = (float*)d_out;

  // Workspace layout (bytes): ~20.3 MB total
  char* ws = (char*)d_ws;
  size_t o = 0;
  bf16_t* WfT = (bf16_t*)(ws + o); o += (size_t)32  * 256 * 2;
  bf16_t* WgT = (bf16_t*)(ws + o); o += (size_t)32  * 256 * 2;
  bf16_t* WhT = (bf16_t*)(ws + o); o += (size_t)256 * 256 * 2;
  bf16_t* WoT = (bf16_t*)(ws + o); o += (size_t)256 * 256 * 2;
  bf16_t* fB  = (bf16_t*)(ws + o); o += (size_t)32768 * 32 * 2;
  bf16_t* gB  = (bf16_t*)(ws + o); o += (size_t)32768 * 32 * 2;
  bf16_t* hhT = (bf16_t*)(ws + o); o += (size_t)8 * 256 * 4096 * 2;

  // Stage 0: weight transposes to bf16 [Cout][Cin]
  k_transpose_bf16<<<dim3((256 * 32  + 255) / 256), dim3(256), 0, stream>>>(Wf, WfT, 256, 32);
  k_transpose_bf16<<<dim3((256 * 32  + 255) / 256), dim3(256), 0, stream>>>(Wg, WgT, 256, 32);
  k_transpose_bf16<<<dim3((256 * 256 + 255) / 256), dim3(256), 0, stream>>>(Wh, WhT, 256, 256);
  k_transpose_bf16<<<dim3((256 * 256 + 255) / 256), dim3(256), 0, stream>>>(Wo, WoT, 256, 256);

  // Stage 1: 2048 row tiles x 20 col tiles = 40960 waves / 4 per block
  k_proj<<<dim3(10240), dim3(128), 0, stream>>>(x, bf, bg, bh, WfT, WgT, WhT, fB, gB, hhT);

  // Stage 2: 8 batches x 256 query tiles = 2048 waves / 4 per block
  k_attn<<<dim3(512), dim3(128), 0, stream>>>(x, fB, gB, hhT, WoT, bo, gamma, out);
}
